// Dynamic_System_34789235097696
// MI455X (gfx1250) — compile-verified, matrix operand order checked by reading, not run on hardware
//
#include <hip/hip_runtime.h>
#include <hip/hip_bf16.h>
#include <math.h>

#define BATCH 32768
#define NH    512

typedef __attribute__((ext_vector_type(16))) __bf16 v16bf;
typedef __attribute__((ext_vector_type(8)))  __bf16 v8bf;
typedef __attribute__((ext_vector_type(8)))  float  v8f;
typedef __attribute__((ext_vector_type(4)))  int    v4i;

union Frag16 { v16bf v; v8bf h[2]; };

#if defined(__has_builtin)
# if __has_builtin(__builtin_amdgcn_global_load_async_to_lds_b128)
#  define USE_ASYNC_LDS 1
# else
#  define USE_ASYNC_LDS 0
# endif
#else
# define USE_ASYNC_LDS 0
#endif

typedef __attribute__((address_space(1))) v4i gv4i;
typedef __attribute__((address_space(3))) v4i lv4i;

__device__ __forceinline__ void stage16(const __bf16* g, __bf16* l) {
#if USE_ASYNC_LDS
  __builtin_amdgcn_global_load_async_to_lds_b128(
      (gv4i*)(void*)g, (lv4i*)(void*)l, 0, 0);
#else
  *(v8bf*)l = *(const v8bf*)g;
#endif
}

__device__ __forceinline__ void wait_stage() {
#if USE_ASYNC_LDS
# if __has_builtin(__builtin_amdgcn_s_wait_asynccnt)
  __builtin_amdgcn_s_wait_asynccnt(0);
# else
  asm volatile("s_wait_asynccnt 0x0" ::: "memory");
# endif
#endif
}

__device__ __forceinline__ __bf16 f2bf(float f) {
  union { float f; unsigned u; } v; v.f = f;
  unsigned r = v.u + 0x7FFFu + ((v.u >> 16) & 1u);
  union { unsigned short s; __bf16 b; } o; o.s = (unsigned short)(r >> 16);
  return o.b;
}
__device__ __forceinline__ float bf2f(__bf16 b) {
  union { unsigned short s; __bf16 b; } i; i.b = b;
  union { unsigned u; float f; } o; o.u = ((unsigned)i.s) << 16;
  return o.f;
}

enum { MODE_H2 = 0, MODE_F32 = 1, MODE_UDOT2 = 2, MODE_UDOT1 = 3, MODE_U1 = 4 };

#define GM 256
#define GN 64
#define GK 32
#define LPAD 40   // padded LDS row (elements); 80B row keeps 16B alignment, staggers banks

// C = A[B,512](bf16 row-major) @ Wn (bf16, n-major: Wn[n*512+k]) with fused epilogues.
// Block tile 256x64; each of 8 waves computes 32(M) x 64(N) = 8 WMMA tiles per k-step.
__global__ __launch_bounds__(256)
void gemm_bf16_512(const __bf16* __restrict__ A,
                   const __bf16* __restrict__ Wn,
                   int mode,
                   const float*  __restrict__ bias,
                   const float*  __restrict__ w3,
                   const __bf16* __restrict__ h2buf,
                   const __bf16* __restrict__ h1buf,
                   const float*  __restrict__ wubuf,
                   const __bf16* __restrict__ hdbuf,
                   __bf16* __restrict__ outb,
                   __bf16* __restrict__ outb2,
                   float*  __restrict__ outf)
{
  __shared__ __bf16 As[GM][LPAD];  // A tile, m-major (20 KB)
  __shared__ __bf16 Bt[GN][LPAD];  // B tile, n-major (5 KB)
  const int tid  = threadIdx.x;
  const int lane = tid & 31;
  const int wave = tid >> 5;
  const int m0 = blockIdx.x * GM;
  const int n0 = blockIdx.y * GN;

  v8f acc[8] = {};

  for (int k0 = 0; k0 < NH; k0 += GK) {
    // Stage A tile: each thread copies its own 64B row (4 x 16B)
    #pragma unroll
    for (int c = 0; c < 4; ++c)
      stage16(A + (size_t)(m0 + tid) * NH + k0 + c * 8, &As[tid][c * 8]);
    { // Stage B tile (n-major -> n-major LDS): 1 x 16B per thread
      int n  = tid >> 2;
      int c8 = (tid & 3) * 8;
      stage16(Wn + (size_t)(n0 + n) * NH + k0 + c8, &Bt[n][c8]);
    }
    if (tid == 0 && k0 + GK < NH)
      __builtin_prefetch(Wn + (size_t)n0 * NH + k0 + GK, 0, 1);
    wait_stage();
    __syncthreads();

    // Two A fragments (rows wave*32 .. +31); ISA 16-bit A 16x32 layout
    const int kh = (lane >> 4) * 8;
    const int rbase = wave * 32 + (lane & 15);
    Frag16 af0, af1;
    af0.h[0] = *(const v8bf*)&As[rbase][kh];
    af0.h[1] = *(const v8bf*)&As[rbase][16 + kh];
    af1.h[0] = *(const v8bf*)&As[rbase + 16][kh];
    af1.h[1] = *(const v8bf*)&As[rbase + 16][16 + kh];
    #pragma unroll
    for (int t = 0; t < 4; ++t) {
      Frag16 bfr;
      int ncol = t * 16 + (lane & 15);
      bfr.h[0] = *(const v8bf*)&Bt[ncol][kh];
      bfr.h[1] = *(const v8bf*)&Bt[ncol][16 + kh];
      acc[t] = __builtin_amdgcn_wmma_f32_16x16x32_bf16(
          false, af0.v, false, bfr.v, (short)0, acc[t], false, false);
      acc[4 + t] = __builtin_amdgcn_wmma_f32_16x16x32_bf16(
          false, af1.v, false, bfr.v, (short)0, acc[4 + t], false, false);
    }
    __syncthreads();
  }

  // Epilogue; C layout: lane<16 -> N=lane, M=r ; lane>=16 -> N=lane-16, M=8+r
  #pragma unroll
  for (int half = 0; half < 2; ++half) {
    const int mbase = m0 + wave * 32 + half * 16 + ((lane >> 4) * 8);
    #pragma unroll
    for (int t = 0; t < 4; ++t) {
      int n = n0 + t * 16 + (lane & 15);
      #pragma unroll
      for (int r = 0; r < 8; ++r) {
        size_t idx = (size_t)(mbase + r) * NH + n;
        float c = acc[half * 4 + t][r];
        if (mode == MODE_H2) {
          float h = tanhf(c + bias[n]);
          outb[idx] = f2bf(h);
          if (outb2) outb2[idx] = f2bf(w3[n] * (1.f - h * h));
        } else if (mode == MODE_F32) {
          outf[idx] = c;
        } else if (mode == MODE_UDOT2) {
          float h2 = bf2f(h2buf[idx]);
          outb[idx] = f2bf(w3[n] * (-2.f * h2 * (1.f - h2 * h2)) * c);
        } else if (mode == MODE_UDOT1) {
          float h1 = bf2f(h1buf[idx]);
          outf[idx] = c * (1.f - h1 * h1)
                    + wubuf[idx] * (-2.f * h1) * bf2f(hdbuf[idx]);
        } else { // MODE_U1
          float h1 = bf2f(h1buf[idx]);
          outf[idx] = c * (1.f - h1 * h1);
        }
      }
    }
  }
}

// h1 = tanh(concat(x0,x1,x2) @ W1 + b1), stored bf16
__global__ void layer1_kernel(const float* __restrict__ x0, int w0,
                              const float* __restrict__ x1, int w1,
                              const float* __restrict__ x2, int w2,
                              const float* __restrict__ W1,
                              const float* __restrict__ b1,
                              __bf16* __restrict__ h1)
{
  int idx = blockIdx.x * blockDim.x + threadIdx.x;
  if (idx >= BATCH * NH) return;
  int b = idx >> 9;
  int n = idx & (NH - 1);
  float a = b1[n];
  int k = 0;
  for (int j = 0; j < w0; ++j, ++k) a += x0[b * w0 + j] * W1[k * NH + n];
  for (int j = 0; j < w1; ++j, ++k) a += x1[b * w1 + j] * W1[k * NH + n];
  if (x2)
    for (int j = 0; j < w2; ++j, ++k) a += x2[b * w2 + j] * W1[k * NH + n];
  h1[idx] = f2bf(tanhf(a));
}

// hd = (1 - h1^2) * W1[i,:]  (tangent of layer-1 activation for direction i)
__global__ void hd_kernel(const __bf16* __restrict__ h1,
                          const float* __restrict__ w1row,
                          __bf16* __restrict__ out)
{
  int idx = blockIdx.x * blockDim.x + threadIdx.x;
  if (idx >= BATCH * NH) return;
  int n = idx & (NH - 1);
  float h = bf2f(h1[idx]);
  out[idx] = f2bf((1.f - h * h) * w1row[n]);
}

// out[b*ostride + j*cstride + ooff] = sum_k W1[j,k] * u[b,k], j=0..2
__global__ void reduce_w1_kernel(const float* __restrict__ u,
                                 const float* __restrict__ W1,
                                 float* __restrict__ out,
                                 int ostride, int cstride, int ooff)
{
  int idx = blockIdx.x * blockDim.x + threadIdx.x;
  if (idx >= BATCH * 3) return;
  int b = idx / 3, j = idx % 3;
  const float* up = u  + (size_t)b * NH;
  const float* wp = W1 + (size_t)j * NH;
  float s = 0.f;
  for (int k = 0; k < NH; ++k) s += up[k] * wp[k];
  out[(size_t)b * ostride + j * cstride + ooff] = s;
}

// Bn output: out[b,j] = h2[b,:] @ W3[:,j] + b3[j]
__global__ void bn_out_kernel(const __bf16* __restrict__ h2,
                              const float* __restrict__ W3,
                              const float* __restrict__ b3,
                              float* __restrict__ out)
{
  int idx = blockIdx.x * blockDim.x + threadIdx.x;
  if (idx >= BATCH * 3) return;
  int b = idx / 3, j = idx % 3;
  float s = b3[j];
  const __bf16* hp = h2 + (size_t)b * NH;
  for (int k = 0; k < NH; ++k) s += bf2f(hp[k]) * W3[k * 3 + j];
  out[idx] = s;
}

// Per-sample 3x3 solve: out = inv(H) @ (tau + bout + g)
__global__ void solve_kernel(const float* __restrict__ H,
                             const float* __restrict__ tau,
                             const float* __restrict__ bout,
                             const float* __restrict__ g,
                             float* __restrict__ out)
{
  int b = blockIdx.x * blockDim.x + threadIdx.x;
  if (b >= BATCH) return;
  const float* h = H + (size_t)b * 9;
  float a00=h[0],a01=h[1],a02=h[2],a10=h[3],a11=h[4],a12=h[5],a20=h[6],a21=h[7],a22=h[8];
  float r0 = tau[b*3+0] + bout[b*3+0] + g[b*3+0];
  float r1 = tau[b*3+1] + bout[b*3+1] + g[b*3+1];
  float r2 = tau[b*3+2] + bout[b*3+2] + g[b*3+2];
  float c00=a11*a22-a12*a21, c01=a02*a21-a01*a22, c02=a01*a12-a02*a11;
  float c10=a12*a20-a10*a22, c11=a00*a22-a02*a20, c12=a02*a10-a00*a12;
  float c20=a10*a21-a11*a20, c21=a01*a20-a00*a21, c22=a00*a11-a01*a10;
  float det = a00*c00 + a01*c10 + a02*c20;
  float inv = 1.f / det;
  out[b*3+0] = (c00*r0 + c01*r1 + c02*r2) * inv;
  out[b*3+1] = (c10*r0 + c11*r1 + c12*r2) * inv;
  out[b*3+2] = (c20*r0 + c21*r1 + c22*r2) * inv;
}

// Wfwd[n*512+k] = bf16(W[k,n])  (n-major B for X@W)
// Wbwd[n*512+k] = bf16(W[n,k])  (n-major B for X@W^T == straight copy)
__global__ void convert_w_kernel(const float* __restrict__ W,
                                 __bf16* __restrict__ Wfwd,
                                 __bf16* __restrict__ Wbwd)
{
  int idx = blockIdx.x * blockDim.x + threadIdx.x;
  if (idx >= NH * NH) return;
  int k = idx >> 9, n = idx & (NH - 1);
  __bf16 v = f2bf(W[idx]);          // W[k][n]
  Wfwd[(size_t)n * NH + k] = v;     // transpose into n-major
  if (Wbwd) Wbwd[idx] = v;          // row-major copy == n-major W^T
}

extern "C" void kernel_launch(void* const* d_in, const int* in_sizes, int n_in,
                              void* d_out, int out_size, void* d_ws, size_t ws_size,
                              hipStream_t stream)
{
  (void)in_sizes; (void)n_in; (void)out_size; (void)ws_size;
  const float* theta_dot = (const float*)d_in[0];
  const float* s_dot     = (const float*)d_in[1];
  const float* theta     = (const float*)d_in[2];
  const float* z         = (const float*)d_in[3];
  const float* s_in      = (const float*)d_in[4];
  const float* s_Ddot    = (const float*)d_in[5];
  const float* tau       = (const float*)d_in[6];
  const float* T_W1 = (const float*)d_in[7];
  const float* T_b1 = (const float*)d_in[8];
  const float* T_W2 = (const float*)d_in[9];
  const float* T_b2 = (const float*)d_in[10];
  const float* T_W3 = (const float*)d_in[11];
  const float* V_W1 = (const float*)d_in[13];
  const float* V_b1 = (const float*)d_in[14];
  const float* V_W2 = (const float*)d_in[15];
  const float* V_b2 = (const float*)d_in[16];
  const float* V_W3 = (const float*)d_in[17];
  const float* B_W1 = (const float*)d_in[19];
  const float* B_b1 = (const float*)d_in[20];
  const float* B_W2 = (const float*)d_in[21];
  const float* B_b2 = (const float*)d_in[22];
  const float* B_W3 = (const float*)d_in[23];
  const float* B_b3 = (const float*)d_in[24];

  char* ws = (char*)d_ws;
  size_t off = 0;
  auto carve = [&](size_t bytes) -> void* {
    void* p = ws + off;
    off += (bytes + 255) & ~(size_t)255;
    return p;
  };
  const size_t NB = (size_t)BATCH * NH;
  __bf16* h1  = (__bf16*)carve(NB * 2);
  __bf16* h2  = (__bf16*)carve(NB * 2);
  __bf16* u2  = (__bf16*)carve(NB * 2);
  __bf16* hd  = (__bf16*)carve(NB * 2);
  __bf16* ud2 = (__bf16*)carve(NB * 2);
  float*  WU  = (float*)carve(NB * 4);
  float*  UD1 = (float*)carve(NB * 4);
  __bf16* WfT = (__bf16*)carve((size_t)NH * NH * 2);  // T fwd (n-major)
  __bf16* WbT = (__bf16*)carve((size_t)NH * NH * 2);  // T bwd
  __bf16* WfV = (__bf16*)carve((size_t)NH * NH * 2);
  __bf16* WbV = (__bf16*)carve((size_t)NH * NH * 2);
  __bf16* WfB = (__bf16*)carve((size_t)NH * NH * 2);
  float* Hm = (float*)carve((size_t)BATCH * 9 * 4);
  float* gv = (float*)carve((size_t)BATCH * 3 * 4);
  float* bo = (float*)carve((size_t)BATCH * 3 * 4);

  dim3 ggrid(BATCH / GM, NH / GN);
  const int EL = BATCH * NH / 256;
  const int R3 = (BATCH * 3 + 255) / 256;
  const int WC = NH * NH / 256;

  convert_w_kernel<<<WC, 256, 0, stream>>>(T_W2, WfT, WbT);
  convert_w_kernel<<<WC, 256, 0, stream>>>(V_W2, WfV, WbV);
  convert_w_kernel<<<WC, 256, 0, stream>>>(B_W2, WfB, nullptr);

  // ===== T network: forward + backward + 3 Hessian JVP directions =====
  layer1_kernel<<<EL, 256, 0, stream>>>(theta_dot, 3, s_dot, 3, nullptr, 0, T_W1, T_b1, h1);
  gemm_bf16_512<<<ggrid, 256, 0, stream>>>(h1, WfT, MODE_H2, T_b2, T_W3,
      nullptr, nullptr, nullptr, nullptr, h2, u2, nullptr);
  gemm_bf16_512<<<ggrid, 256, 0, stream>>>(u2, WbT, MODE_F32, nullptr, nullptr,
      nullptr, nullptr, nullptr, nullptr, nullptr, nullptr, WU);
  for (int i = 0; i < 3; ++i) {
    hd_kernel<<<EL, 256, 0, stream>>>(h1, T_W1 + (size_t)i * NH, hd);
    gemm_bf16_512<<<ggrid, 256, 0, stream>>>(hd, WfT, MODE_UDOT2, nullptr, T_W3,
        h2, nullptr, nullptr, nullptr, ud2, nullptr, nullptr);
    gemm_bf16_512<<<ggrid, 256, 0, stream>>>(ud2, WbT, MODE_UDOT1, nullptr, nullptr,
        nullptr, h1, WU, hd, nullptr, nullptr, UD1);
    reduce_w1_kernel<<<R3, 256, 0, stream>>>(UD1, T_W1, Hm, 9, 3, i);
  }

  // ===== V network: forward + gradient wrt theta =====
  layer1_kernel<<<EL, 256, 0, stream>>>(theta, 3, z, 1, nullptr, 0, V_W1, V_b1, h1);
  gemm_bf16_512<<<ggrid, 256, 0, stream>>>(h1, WfV, MODE_H2, V_b2, V_W3,
      nullptr, nullptr, nullptr, nullptr, h2, u2, nullptr);
  gemm_bf16_512<<<ggrid, 256, 0, stream>>>(u2, WbV, MODE_U1, nullptr, nullptr,
      nullptr, h1, nullptr, nullptr, nullptr, nullptr, UD1);
  reduce_w1_kernel<<<R3, 256, 0, stream>>>(UD1, V_W1, gv, 3, 1, 0);

  // ===== Bn network: plain forward =====
  layer1_kernel<<<EL, 256, 0, stream>>>(theta, 3, s_in, 3, s_Ddot, 3, B_W1, B_b1, h1);
  gemm_bf16_512<<<ggrid, 256, 0, stream>>>(h1, WfB, MODE_H2, B_b2, nullptr,
      nullptr, nullptr, nullptr, nullptr, h2, nullptr, nullptr);
  bn_out_kernel<<<R3, 256, 0, stream>>>(h2, B_W3, B_b3, bo);

  // ===== assemble rhs and solve 3x3 per sample =====
  solve_kernel<<<BATCH / 256, 256, 0, stream>>>(Hm, tau, bo, gv, (float*)d_out);
}
